// BigBirdRegressor_MLP_cat_Reduced_42150809043592
// MI455X (gfx1250) — matrix operand order repaired and checked by reading, not compiled, not run on hardware
//
#include <hip/hip_runtime.h>

typedef __bf16 bf16;
typedef __attribute__((ext_vector_type(16))) __bf16    v16bf;
typedef __attribute__((ext_vector_type(8)))  float     v8f;
typedef __attribute__((ext_vector_type(4)))  unsigned  v4u;

#define WMMA_BF16(a, b, c) \
  __builtin_amdgcn_wmma_f32_16x16x32_bf16(false, (a), false, (b), (short)0, (c), false, false)

// ---------------------------------------------------------------------------
// CDNA5 async global->LDS copy (GLOBAL_LOAD_ASYNC_TO_LDS_B128, ASYNCcnt).
// Builtins are absent on this toolchain (probed via __has_builtin last round),
// so use inline asm per CDNA5 ISA:
//   each lane copies 16 bytes from its VADDR to LDS at its VDST offset; wait with s_wait_asynccnt.
// Generic LDS pointers hold the LDS byte offset in their low 32 bits.
// ---------------------------------------------------------------------------
__device__ inline void copy16_to_lds(const bf16* g, unsigned short* l) {
  unsigned loff = (unsigned)(size_t)(void*)l;
  asm volatile("global_load_async_to_lds_b128 %0, %1, off"
               :: "v"(loff), "v"(g)
               : "memory");
}

__device__ inline void copy_wait() {
  asm volatile("s_wait_asynccnt 0x0" ::: "memory");
}

// ---- model constants ----
constexpr int B_   = 4;
constexpr int S_   = 4096;
constexpr int H_   = 768;
constexpr int NH_  = 12;
constexpr int D_   = 64;
constexpr int BLK_ = 64;
constexpr int NBLK = S_ / BLK_;       // 64
constexpr int FF_  = 4 * H_;          // 3072
constexpr int MTOK = B_ * S_;         // 16384
constexpr int H1_  = 1000;
constexpr int H2_  = 500;
__device__ constexpr float LN_EPS = 1e-12f;
__device__ constexpr float BN_S   = 0.9999950000374996f;  // 1/sqrt(1+1e-5)

union Frag {
  unsigned u[8];
  v4u      q[2];
  v16bf    v;
};

__device__ inline v8f v8set(float x) {
  v8f r;
#pragma unroll
  for (int i = 0; i < 8; ++i) r[i] = x;
  return r;
}

__device__ inline unsigned short bf16bits(float f) {
  union { bf16 h; unsigned short u; } c;
  c.h = (bf16)f;
  return c.u;
}

// ---------------------------------------------------------------------------
// f32 -> bf16 convert + WMMA-B-fragment swizzle.
// dst layout: [(K/32)][N][32]: element (k,n) -> ((k>>5)*N + n)*32 + ((k>>4)&1)*16 + (k&15)
// ---------------------------------------------------------------------------
__global__ void cvt_swizzle_kernel(const float* __restrict__ in, bf16* __restrict__ out,
                                   int K, int N) {
  int i = blockIdx.x * 256 + threadIdx.x;
  if (i >= K * N) return;
  int k = i / N, n = i % N;
  size_t dst = ((size_t)(k >> 5) * N + n) * 32 + ((k >> 4) & 1) * 16 + (k & 15);
  out[dst] = (bf16)in[i];
}

// ---------------------------------------------------------------------------
// Embedding gather + LayerNorm  (one block per token row, 256 thr, H=768)
// ---------------------------------------------------------------------------
__global__ __launch_bounds__(256)
void embed_ln_kernel(const int* __restrict__ ids, const float* __restrict__ we,
                     const float* __restrict__ pe, const float* __restrict__ te,
                     const float* __restrict__ g, const float* __restrict__ bta,
                     float* __restrict__ outf, bf16* __restrict__ outh) {
  __shared__ float red[256];
  const int row = blockIdx.x;            // b*S + s
  const int s   = row & (S_ - 1);
  const int tid = threadIdx.x;
  const int id  = ids[row];
  float vals[3], sum = 0.f, sq = 0.f;
#pragma unroll
  for (int i = 0; i < 3; ++i) {
    int j = tid + i * 256;
    float v = we[id * H_ + j] + pe[(size_t)s * H_ + j] + te[j];
    vals[i] = v; sum += v; sq += v * v;
  }
  red[tid] = sum; __syncthreads();
  for (int st = 128; st > 0; st >>= 1) { if (tid < st) red[tid] += red[tid + st]; __syncthreads(); }
  float mean = red[0] * (1.f / H_); __syncthreads();
  red[tid] = sq; __syncthreads();
  for (int st = 128; st > 0; st >>= 1) { if (tid < st) red[tid] += red[tid + st]; __syncthreads(); }
  float var = red[0] * (1.f / H_) - mean * mean;
  float inv = rsqrtf(var + LN_EPS);
#pragma unroll
  for (int i = 0; i < 3; ++i) {
    int j = tid + i * 256;
    float y = (vals[i] - mean) * inv * g[j] + bta[j];
    outf[(size_t)row * H_ + j] = y;
    outh[(size_t)row * H_ + j] = (bf16)y;
  }
}

// ---------------------------------------------------------------------------
// LayerNorm (f32 in -> f32 + bf16 out)
// ---------------------------------------------------------------------------
__global__ __launch_bounds__(256)
void layernorm_kernel(const float* __restrict__ in, const float* __restrict__ g,
                      const float* __restrict__ bta, float* __restrict__ outf,
                      bf16* __restrict__ outh) {
  __shared__ float red[256];
  const int row = blockIdx.x;
  const int tid = threadIdx.x;
  const float* x = in + (size_t)row * H_;
  float vals[3], sum = 0.f, sq = 0.f;
#pragma unroll
  for (int i = 0; i < 3; ++i) {
    float v = x[tid + i * 256];
    vals[i] = v; sum += v; sq += v * v;
  }
  red[tid] = sum; __syncthreads();
  for (int st = 128; st > 0; st >>= 1) { if (tid < st) red[tid] += red[tid + st]; __syncthreads(); }
  float mean = red[0] * (1.f / H_); __syncthreads();
  red[tid] = sq; __syncthreads();
  for (int st = 128; st > 0; st >>= 1) { if (tid < st) red[tid] += red[tid + st]; __syncthreads(); }
  float var = red[0] * (1.f / H_) - mean * mean;
  float inv = rsqrtf(var + LN_EPS);
#pragma unroll
  for (int i = 0; i < 3; ++i) {
    int j = tid + i * 256;
    float y = (vals[i] - mean) * inv * g[j] + bta[j];
    outf[(size_t)row * H_ + j] = y;
    outh[(size_t)row * H_ + j] = (bf16)y;
  }
}

// ---------------------------------------------------------------------------
// WMMA GEMM: C[M,N] = (A[M,K](bf16) @ Wsw[K,N](bf16 swizzled) + bias) * oscale
// Block tile 128x128, K-step 64, 8 waves, each wave 64x32 (4x2 WMMA tiles).
// A double-buffered in LDS via async global->LDS DMA; B fragments loaded
// straight from global (L2-resident swizzled weights, 2x b128/frag).
// EPI: 1 = tanh-GELU -> bf16 [M,N]
//      2 = +residual(f32) -> f32 [M,N]
//      3 = bf16 -> [B,NH,S,D] (Q,K; oscale folds softmax 1/sqrt(D) into Q)
//      4 = bf16 -> [B,NH,D,S] (V transposed)
// ---------------------------------------------------------------------------
template <int EPI>
__global__ __launch_bounds__(256)
void gemm_wmma_kernel(const bf16* __restrict__ A, const bf16* __restrict__ Wsw,
                      const float* __restrict__ bias, const float* __restrict__ resid,
                      void* __restrict__ out, float oscale, int M, int N, int K) {
  __shared__ unsigned short As[2][128 * 72];   // [m][k], stride 72, ping-pong
  const int tid  = threadIdx.x;
  const int lane = tid & 31;
  const int wv   = tid >> 5;
  const int wr   = wv >> 2;   // 0..1
  const int wc   = wv & 3;    // 0..3
  const int m0   = blockIdx.y * 128;
  const int n0   = blockIdx.x * 128;

  v8f acc[4][2];
#pragma unroll
  for (int mi = 0; mi < 4; ++mi)
#pragma unroll
    for (int ni = 0; ni < 2; ++ni) acc[mi][ni] = v8set(0.f);

  auto issueA = [&](int k0, int buf) {
#pragma unroll
    for (int c = 0; c < 4; ++c) {   // A tile: 128x64 = 1024 8-elem chunks
      int chunk = tid + c * 256;
      int row = chunk >> 3, col8 = chunk & 7;
      copy16_to_lds(A + (size_t)(m0 + row) * K + k0 + col8 * 8,
                    &As[buf][row * 72 + col8 * 8]);
    }
  };

  const int nk = K >> 6;
  issueA(0, 0);
  for (int i = 0; i < nk; ++i) {
    copy_wait();
    __syncthreads();
    if (i + 1 < nk) issueA((i + 1) << 6, (i + 1) & 1);
    const unsigned short* Ab = &As[i & 1][0];
    const int k0 = i << 6;
#pragma unroll
    for (int ks = 0; ks < 2; ++ks) {
      // B fragments first (global, long latency): 2x global_load_b128 each
      Frag bfr[2];
      int kb32 = (k0 >> 5) + ks;
#pragma unroll
      for (int ni = 0; ni < 2; ++ni) {
        int n = n0 + wc * 32 + ni * 16 + (lane & 15);
        const bf16* wp = Wsw + ((size_t)kb32 * N + n) * 32 + ((lane >> 4) << 4);
        bfr[ni].q[0] = *(const v4u*)(wp);
        bfr[ni].q[1] = *(const v4u*)(wp + 8);
      }
      // A fragments: 2x ds_load_b128 each
      Frag af[4];
#pragma unroll
      for (int mi = 0; mi < 4; ++mi) {
        int Mrow = wr * 64 + mi * 16 + (lane & 15);
        int kb   = ks * 32 + ((lane >> 4) << 4);
        af[mi].q[0] = *(const v4u*)(&Ab[Mrow * 72 + kb]);
        af[mi].q[1] = *(const v4u*)(&Ab[Mrow * 72 + kb + 8]);
      }
#pragma unroll
      for (int mi = 0; mi < 4; ++mi)
#pragma unroll
        for (int ni = 0; ni < 2; ++ni)
          acc[mi][ni] = WMMA_BF16(af[mi].v, bfr[ni].v, acc[mi][ni]);
    }
    __syncthreads();
  }

  // epilogue
#pragma unroll
  for (int ni = 0; ni < 2; ++ni) {
    int col = n0 + wc * 32 + ni * 16 + (lane & 15);
    float bv = bias[col];
#pragma unroll
    for (int mi = 0; mi < 4; ++mi) {
#pragma unroll
      for (int r = 0; r < 8; ++r) {
        int row = m0 + wr * 64 + mi * 16 + r + ((lane >> 4) << 3);
        float v = (acc[mi][ni][r] + bv) * oscale;
        if (EPI == 1) {
          float gl = 0.5f * v * (1.f + tanhf(0.7978845608028654f * (v + 0.044715f * v * v * v)));
          ((bf16*)out)[(size_t)row * N + col] = (bf16)gl;
        } else if (EPI == 2) {
          ((float*)out)[(size_t)row * N + col] = v + resid[(size_t)row * N + col];
        } else if (EPI == 3) {
          int b = row >> 12, s = row & 4095;
          int h = col >> 6,  d = col & 63;
          ((bf16*)out)[(((size_t)b * NH_ + h) * S_ + s) * D_ + d] = (bf16)v;
        } else {  // EPI == 4: V transposed [B,NH,D,S]
          int b = row >> 12, s = row & 4095;
          int h = col >> 6,  d = col & 63;
          ((bf16*)out)[(((size_t)b * NH_ + h) * D_ + d) * S_ + s] = (bf16)v;
        }
      }
    }
  }
}

// ---------------------------------------------------------------------------
// Fused BigBird block attention. One workgroup (4 waves) per (b, h, q-block).
// q (pre-scaled by 1/sqrt(D)),k: bf16 [B,NH,S,D]; vT: bf16 [B,NH,D,S];
// out: bf16 [B,S,H]. Flash-style online softmax in C-fragment layout,
// P converted C->A layout via per-wave LDS strip. k/v tiles double-buffered
// through the async global->LDS DMA.
// ---------------------------------------------------------------------------
__global__ __launch_bounds__(128)
void attn_kernel(const bf16* __restrict__ q, const bf16* __restrict__ k,
                 const bf16* __restrict__ vT, const int* __restrict__ rand_idx,
                 bf16* __restrict__ out) {
  __shared__ unsigned short kt[2][64 * 72];  // [key][d], ping-pong
  __shared__ unsigned short vt[2][64 * 72];  // [d][key], ping-pong
  __shared__ unsigned short pt[64 * 72];     // [qrow][key], per-wave strips
  const int qb = blockIdx.x, h = blockIdx.y, b = blockIdx.z;
  const int tid = threadIdx.x, lane = tid & 31, wv = tid >> 5;
  const size_t headoff = ((size_t)b * NH_ + h) * S_ * D_;
  const bf16* qp  = q  + headoff + (size_t)qb * BLK_ * D_;
  const bf16* kp  = k  + headoff;
  const bf16* vtp = vT + headoff;            // [d][s]

  // q A-fragments for both K-steps, loaded straight from global (b128 pairs)
  Frag qa[2];
  {
    int Mrow = wv * 16 + (lane & 15);
#pragma unroll
    for (int ks = 0; ks < 2; ++ks) {
      int kb0 = ks * 32 + ((lane >> 4) << 4);
      qa[ks].q[0] = *(const v4u*)(qp + Mrow * D_ + kb0);
      qa[ks].q[1] = *(const v4u*)(qp + Mrow * D_ + kb0 + 8);
    }
  }

  int list[8];
  const bool full = (qb == 0 || qb == NBLK - 1);
  int nkb;
  if (full) {
    nkb = NBLK;
  } else {
    list[0] = qb - 1; list[1] = qb; list[2] = qb + 1;
    list[3] = 0;      list[4] = NBLK - 1;
    list[5] = rand_idx[(qb - 1) * 3 + 0];
    list[6] = rand_idx[(qb - 1) * 3 + 1];
    list[7] = rand_idx[(qb - 1) * 3 + 2];
    nkb = 8;
  }

  auto issueKV = [&](int kb, int buf) {
#pragma unroll
    for (int c = 0; c < 4; ++c) {
      int chunk = tid + c * 128;
      int row = chunk >> 3, col8 = chunk & 7;
      copy16_to_lds(kp + ((size_t)kb * BLK_ + row) * D_ + col8 * 8,
                    &kt[buf][row * 72 + col8 * 8]);
      copy16_to_lds(vtp + (size_t)row * S_ + kb * BLK_ + col8 * 8,
                    &vt[buf][row * 72 + col8 * 8]);
    }
  };

  v8f m = v8set(-1e30f), l = v8set(0.f);
  v8f o[4];
#pragma unroll
  for (int dt = 0; dt < 4; ++dt) o[dt] = v8set(0.f);

  issueKV(full ? 0 : list[0], 0);
  for (int it = 0; it < nkb; ++it) {
    copy_wait();
    __syncthreads();
    if (it + 1 < nkb) issueKV(full ? (it + 1) : list[it + 1], (it + 1) & 1);
    const unsigned short* ktb = &kt[it & 1][0];
    const unsigned short* vtb = &vt[it & 1][0];

    // scores: 16x64 per wave = 4 column tiles (q pre-scaled, no extra VALU)
    v8f s[4];
#pragma unroll
    for (int ct = 0; ct < 4; ++ct) s[ct] = v8set(0.f);
#pragma unroll
    for (int ks = 0; ks < 2; ++ks) {
      Frag bk[4];
      int db = ks * 32 + ((lane >> 4) << 4);
#pragma unroll
      for (int ct = 0; ct < 4; ++ct) {
        int Ncol = ct * 16 + (lane & 15);
        bk[ct].q[0] = *(const v4u*)(&ktb[Ncol * 72 + db]);
        bk[ct].q[1] = *(const v4u*)(&ktb[Ncol * 72 + db + 8]);
      }
#pragma unroll
      for (int ct = 0; ct < 4; ++ct)
        s[ct] = WMMA_BF16(qa[ks].v, bk[ct].v, s[ct]);
    }

    // per-row max over this key block (halves stay independent: masks 1,2,4,8)
    v8f mt;
#pragma unroll
    for (int r = 0; r < 8; ++r)
      mt[r] = fmaxf(fmaxf(s[0][r], s[1][r]), fmaxf(s[2][r], s[3][r]));
#pragma unroll
    for (int msk = 1; msk <= 8; msk <<= 1)
#pragma unroll
      for (int r = 0; r < 8; ++r) mt[r] = fmaxf(mt[r], __shfl_xor(mt[r], msk, 32));

    v8f mn, corr;
#pragma unroll
    for (int r = 0; r < 8; ++r) {
      mn[r] = fmaxf(m[r], mt[r]);
      corr[r] = __expf(m[r] - mn[r]);
    }

    v8f ps = v8set(0.f);
#pragma unroll
    for (int ct = 0; ct < 4; ++ct) {
#pragma unroll
      for (int r = 0; r < 8; ++r) {
        float e = __expf(s[ct][r] - mn[r]);
        ps[r] += e;
        int row = wv * 16 + r + ((lane >> 4) << 3);
        int col = ct * 16 + (lane & 15);
        pt[row * 72 + col] = bf16bits(e);
      }
    }
#pragma unroll
    for (int msk = 1; msk <= 8; msk <<= 1)
#pragma unroll
      for (int r = 0; r < 8; ++r) ps[r] += __shfl_xor(ps[r], msk, 32);

#pragma unroll
    for (int r = 0; r < 8; ++r) {
      l[r] = l[r] * corr[r] + ps[r];
      m[r] = mn[r];
    }
#pragma unroll
    for (int dt = 0; dt < 4; ++dt)
#pragma unroll
      for (int r = 0; r < 8; ++r) o[dt][r] *= corr[r];

    // O += P @ V  (same-wave LDS RAW on pt: DS ops in-order per wave)
#pragma unroll
    for (int ks2 = 0; ks2 < 2; ++ks2) {
      Frag pa;
      int Mrow = wv * 16 + (lane & 15);
      int kb0  = ks2 * 32 + ((lane >> 4) << 4);
      pa.q[0] = *(const v4u*)(&pt[Mrow * 72 + kb0]);
      pa.q[1] = *(const v4u*)(&pt[Mrow * 72 + kb0 + 8]);
      Frag bv[4];
      int keyb = ks2 * 32 + ((lane >> 4) << 4);
#pragma unroll
      for (int dt = 0; dt < 4; ++dt) {
        int Ncol = dt * 16 + (lane & 15);    // = d
        bv[dt].q[0] = *(const v4u*)(&vtb[Ncol * 72 + keyb]);
        bv[dt].q[1] = *(const v4u*)(&vtb[Ncol * 72 + keyb + 8]);
      }
#pragma unroll
      for (int dt = 0; dt < 4; ++dt)
        o[dt] = WMMA_BF16(pa.v, bv[dt].v, o[dt]);
    }
  }

  // finalize: O / l  -> out[b, qb*64+row, h*64+col]
  v8f inv;
#pragma unroll
  for (int r = 0; r < 8; ++r) inv[r] = 1.f / l[r];
#pragma unroll
  for (int dt = 0; dt < 4; ++dt) {
#pragma unroll
    for (int r = 0; r < 8; ++r) {
      int row = wv * 16 + r + ((lane >> 4) << 3);
      int col = dt * 16 + (lane & 15);
      out[((size_t)b * S_ + qb * BLK_ + row) * H_ + h * D_ + col] = (bf16)(o[dt][r] * inv[r]);
    }
  }
}

// ---------------------------------------------------------------------------
// reduced[b,s] = dot(x[b,s,:], w_red) + b_red   (one wave per row)
// ---------------------------------------------------------------------------
__global__ __launch_bounds__(256)
void reduce_head_kernel(const float* __restrict__ x, const float* __restrict__ w,
                        const float* __restrict__ brd, float* __restrict__ out) {
  int row  = blockIdx.x * 8 + (threadIdx.x >> 5);
  int lane = threadIdx.x & 31;
  const float* xr = x + (size_t)row * H_;
  float s = 0.f;
  for (int j = lane; j < H_; j += 32) s += xr[j] * w[j];
#pragma unroll
  for (int msk = 16; msk > 0; msk >>= 1) s += __shfl_xor(s, msk, 32);
  if (lane == 0) out[row] = s + brd[0];
}

// ---------------------------------------------------------------------------
// tiny MLP head (eval-mode batchnorm folded into scale)
// ---------------------------------------------------------------------------
__global__ void mlp1_kernel(const float* __restrict__ red, const float* __restrict__ w1,
                            const float* __restrict__ b1, const float* __restrict__ g,
                            const float* __restrict__ bb, float* __restrict__ h1) {
  int idx = blockIdx.x * 256 + threadIdx.x;
  if (idx >= B_ * H1_) return;
  int b = idx / H1_, j = idx % H1_;
  float acc = b1[j];
  const float* r = red + b * S_;
  for (int s = 0; s < S_; ++s) acc += r[s] * w1[(size_t)s * H1_ + j];
  float x = acc * BN_S * g[j] + bb[j];
  h1[idx] = fmaxf(x, 0.f);
}

__global__ void mlp2_kernel(const float* __restrict__ h1, const float* __restrict__ w2,
                            const float* __restrict__ b2, const float* __restrict__ g,
                            const float* __restrict__ bb, float* __restrict__ h2) {
  int idx = blockIdx.x * 256 + threadIdx.x;
  if (idx >= B_ * H2_) return;
  int b = idx / H2_, j = idx % H2_;
  float acc = b2[j];
  const float* hp = h1 + b * H1_;
  for (int i = 0; i < H1_; ++i) acc += hp[i] * w2[(size_t)i * H2_ + j];
  float x = acc * BN_S * g[j] + bb[j];
  h2[idx] = fmaxf(x, 0.f);
}

__global__ void mlp3_kernel(const float* __restrict__ h2, const float* __restrict__ w3,
                            const float* __restrict__ b3, float* __restrict__ logits) {
  int b = threadIdx.x;
  if (b < B_) {
    float acc = b3[0];
    for (int i = 0; i < H2_; ++i) acc += h2[b * H2_ + i] * w3[i];
    logits[b] = acc;
  }
}

// ---------------------------------------------------------------------------
// host
// ---------------------------------------------------------------------------
extern "C" void kernel_launch(void* const* d_in, const int* in_sizes, int n_in,
                              void* d_out, int out_size, void* d_ws, size_t ws_size,
                              hipStream_t stream) {
  (void)in_sizes; (void)n_in; (void)out_size; (void)ws_size;

  const int*   input_ids = (const int*)d_in[0];
  const int*   rand_idx  = (const int*)d_in[1];
  const float* word_emb  = (const float*)d_in[2];
  const float* pos_emb   = (const float*)d_in[3];
  const float* tok_emb   = (const float*)d_in[4];
  const float* ln_emb_g  = (const float*)d_in[5];
  const float* ln_emb_b  = (const float*)d_in[6];
  // layers: base 7 + 16*l : wq bq wk bk wv bv wo bo ln1g ln1b wi bi wf bf ln2g ln2b
  const float* w_red = (const float*)d_in[39];
  const float* b_red = (const float*)d_in[40];
  const float* w1    = (const float*)d_in[41];
  const float* b1    = (const float*)d_in[42];
  const float* bn1_g = (const float*)d_in[43];
  const float* bn1_b = (const float*)d_in[44];
  const float* w2    = (const float*)d_in[45];
  const float* b2    = (const float*)d_in[46];
  const float* bn2_g = (const float*)d_in[47];
  const float* bn2_b = (const float*)d_in[48];
  const float* w3    = (const float*)d_in[49];
  const float* b3    = (const float*)d_in[50];

  char* ws = (char*)d_ws;
  size_t off = 0;
  auto alloc = [&](size_t bytes) -> void* {
    void* p = ws + off;
    off = (off + bytes + 255) & ~(size_t)255;
    return p;
  };

  float* xf    = (float*)alloc((size_t)MTOK * H_ * 4);    // residual / LN output (f32)
  bf16*  xb    = (bf16*)alloc((size_t)MTOK * H_ * 2);     // bf16 copy for GEMMs
  float* tmpf  = (float*)alloc((size_t)MTOK * H_ * 4);    // pre-LN (f32)
  bf16*  qbuf  = (bf16*)alloc((size_t)MTOK * H_ * 2);     // [B,NH,S,D]
  bf16*  kbuf  = (bf16*)alloc((size_t)MTOK * H_ * 2);     // [B,NH,S,D]
  bf16*  vbuf  = (bf16*)alloc((size_t)MTOK * H_ * 2);     // [B,NH,D,S] (transposed)
  bf16*  attn  = (bf16*)alloc((size_t)MTOK * H_ * 2);     // [B,S,H]
  bf16*  ffmid = qbuf;                                    // reuse q/k/v/attn region: [MTOK,3072]
  float* h1    = (float*)alloc((size_t)B_ * H1_ * 4);
  float* h2    = (float*)alloc((size_t)B_ * H2_ * 4);

  // bf16 fragment-swizzled weights
  bf16* wbf[2][6];   // wq wk wv wo wi wf
  const int wK[6] = { H_, H_, H_, H_, H_,  FF_ };
  const int wN[6] = { H_, H_, H_, H_, FF_, H_  };
  for (int l = 0; l < 2; ++l)
    for (int wi = 0; wi < 6; ++wi)
      wbf[l][wi] = (bf16*)alloc((size_t)wK[wi] * wN[wi] * 2);

  // --- convert + swizzle weights to bf16 B-fragment layout ---
  for (int l = 0; l < 2; ++l) {
    const int base = 7 + 16 * l;
    const int srcidx[6] = { base + 0, base + 2, base + 4, base + 6, base + 10, base + 12 };
    for (int wi = 0; wi < 6; ++wi) {
      int n = wK[wi] * wN[wi];
      cvt_swizzle_kernel<<<(n + 255) / 256, 256, 0, stream>>>(
          (const float*)d_in[srcidx[wi]], wbf[l][wi], wK[wi], wN[wi]);
    }
  }

  // --- embedding + LN ---
  embed_ln_kernel<<<MTOK, 256, 0, stream>>>(input_ids, word_emb, pos_emb, tok_emb,
                                            ln_emb_g, ln_emb_b, xf, xb);

  const dim3 g768(H_ / 128, MTOK / 128);    // (6,128)
  const dim3 gff(FF_ / 128, MTOK / 128);    // (24,128)
  const float QSCALE = 0.125f;              // 1/sqrt(D), folded into Q projection

  for (int l = 0; l < 2; ++l) {
    const int base = 7 + 16 * l;
    const float* bq   = (const float*)d_in[base + 1];
    const float* bk   = (const float*)d_in[base + 3];
    const float* bv   = (const float*)d_in[base + 5];
    const float* bo   = (const float*)d_in[base + 7];
    const float* ln1g = (const float*)d_in[base + 8];
    const float* ln1b = (const float*)d_in[base + 9];
    const float* bi   = (const float*)d_in[base + 11];
    const float* bfb  = (const float*)d_in[base + 13];
    const float* ln2g = (const float*)d_in[base + 14];
    const float* ln2b = (const float*)d_in[base + 15];

    // QKV projections (Q pre-scaled; Q,K -> [B,NH,S,D]; V -> [B,NH,D,S])
    gemm_wmma_kernel<3><<<g768, 256, 0, stream>>>(xb, wbf[l][0], bq, nullptr, qbuf, QSCALE, MTOK, H_, H_);
    gemm_wmma_kernel<3><<<g768, 256, 0, stream>>>(xb, wbf[l][1], bk, nullptr, kbuf, 1.0f, MTOK, H_, H_);
    gemm_wmma_kernel<4><<<g768, 256, 0, stream>>>(xb, wbf[l][2], bv, nullptr, vbuf, 1.0f, MTOK, H_, H_);

    // BigBird block-sparse attention
    attn_kernel<<<dim3(NBLK, NH_, B_), 128, 0, stream>>>(qbuf, kbuf, vbuf, rand_idx, attn);

    // out-proj + residual -> tmpf (f32), then LN1
    gemm_wmma_kernel<2><<<g768, 256, 0, stream>>>(attn, wbf[l][3], bo, xf, tmpf, 1.0f, MTOK, H_, H_);
    layernorm_kernel<<<MTOK, 256, 0, stream>>>(tmpf, ln1g, ln1b, xf, xb);

    // FFN up (GELU) and down (+residual), then LN2
    gemm_wmma_kernel<1><<<gff, 256, 0, stream>>>(xb, wbf[l][4], bi, nullptr, ffmid, 1.0f, MTOK, FF_, H_);
    gemm_wmma_kernel<2><<<g768, 256, 0, stream>>>(ffmid, wbf[l][5], bfb, xf, tmpf, 1.0f, MTOK, H_, FF_);
    layernorm_kernel<<<MTOK, 256, 0, stream>>>(tmpf, ln2g, ln2b, xf, xb);
  }

  float* out = (float*)d_out;              // [logits(4) | reduced(16384) | embedding(B*S*H)]
  reduce_head_kernel<<<MTOK / 8, 256, 0, stream>>>(xf, w_red, b_red, out + 4);
  hipMemcpyAsync(out + 4 + MTOK, xf, (size_t)MTOK * H_ * 4, hipMemcpyDeviceToDevice, stream);

  mlp1_kernel<<<(B_ * H1_ + 255) / 256, 256, 0, stream>>>(out + 4, w1, b1, bn1_g, bn1_b, h1);
  mlp2_kernel<<<(B_ * H2_ + 255) / 256, 256, 0, stream>>>(h1, w2, b2, bn2_g, bn2_b, h2);
  mlp3_kernel<<<1, 32, 0, stream>>>(h2, w3, b3, out);
}
